// DCNv2_17403207483478
// MI455X (gfx1250) — compile-verified, matrix operand order checked by reading, not run on hardware
//
#include <hip/hip_runtime.h>
#include <hip/hip_bf16.h>

// ---------------- problem constants (from reference) ----------------
#define BATCH 2
#define CH    256      // input channels C
#define HH    96
#define WW    96
#define OUTC  256      // output channels O
#define KHW   3
#define K9    9        // KH*KW
#define PAD   1
#define SPIX  (HH*WW)          // 9216 output pixels (stride1/pad1/dil1 => Ho=H)
#define CK    (CH*K9)          // 2304 GEMM K dimension = 72*32
#define HWSZ  (HH*WW)

typedef __attribute__((ext_vector_type(16))) _Float16 v16h;
typedef __attribute__((ext_vector_type(8)))  _Float16 v8h;
typedef __attribute__((ext_vector_type(8)))  float    v8f;

// -------------------------------------------------------------------
// Kernel 1: deformable bilinear gather -> colT[b][s][k*C + c]  (f16)
// One thread per (b,k,s): bilinear weights computed once, channel loop
// packs 8 channels into one 16B store (global_store_b128).
// -------------------------------------------------------------------
__global__ __launch_bounds__(256) void dcn_gather(
    const float* __restrict__ input,   // [B][C][H][W]
    const float* __restrict__ offset,  // [B][2K][H][W]
    const float* __restrict__ mask,    // [B][K][H][W]
    _Float16* __restrict__ colT)       // [B][S][CK]
{
    int tid = blockIdx.x * blockDim.x + threadIdx.x;
    if (tid >= BATCH * K9 * SPIX) return;
    int s = tid % SPIX;
    int t = tid / SPIX;
    int k = t % K9;
    int b = t / K9;
    int h = s / WW;
    int w = s % WW;

    const size_t offb = (size_t)b * (2 * K9) * SPIX;
    float dy = offset[offb + (size_t)(2 * k)     * SPIX + s];
    float dx = offset[offb + (size_t)(2 * k + 1) * SPIX + s];
    float mk = mask[((size_t)b * K9 + k) * SPIX + s];

    float y = (float)(h - PAD) + (float)(k / KHW) + dy;
    float x = (float)(w - PAD) + (float)(k % KHW) + dx;
    float y0f = floorf(y), x0f = floorf(x);
    float wy = y - y0f,    wx = x - x0f;
    int y0 = (int)y0f, x0 = (int)x0f;
    int y1 = y0 + 1,   x1 = x0 + 1;

    float vy0 = (y0 >= 0 && y0 < HH) ? 1.f : 0.f;
    float vy1 = (y1 >= 0 && y1 < HH) ? 1.f : 0.f;
    float vx0 = (x0 >= 0 && x0 < WW) ? 1.f : 0.f;
    float vx1 = (x1 >= 0 && x1 < WW) ? 1.f : 0.f;
    int iy0 = min(max(y0, 0), HH - 1);
    int iy1 = min(max(y1, 0), HH - 1);
    int ix0 = min(max(x0, 0), WW - 1);
    int ix1 = min(max(x1, 0), WW - 1);
    int i00 = iy0 * WW + ix0;
    int i01 = iy0 * WW + ix1;
    int i10 = iy1 * WW + ix0;
    int i11 = iy1 * WW + ix1;
    // fold validity and mask into the 4 corner weights
    float w00 = (1.f - wy) * (1.f - wx) * vy0 * vx0 * mk;
    float w01 = (1.f - wy) * wx         * vy0 * vx1 * mk;
    float w10 = wy         * (1.f - wx) * vy1 * vx0 * mk;
    float w11 = wy         * wx         * vy1 * vx1 * mk;

    const float* ib = input + (size_t)b * CH * HWSZ;
    _Float16* crow = colT + ((size_t)b * SPIX + s) * CK + (size_t)k * CH;
    #pragma unroll 1
    for (int c0 = 0; c0 < CH; c0 += 8) {
        v8h pack;
        #pragma unroll
        for (int i = 0; i < 8; ++i) {
            const float* p = ib + (size_t)(c0 + i) * HWSZ;
            float v = w00 * p[i00] + w01 * p[i01] + w10 * p[i10] + w11 * p[i11];
            pack[i] = (_Float16)v;
        }
        *(v8h*)(crow + c0) = pack;   // 16B store
    }
}

// -------------------------------------------------------------------
// Kernel 2: repack weight (O,C,3,3) f32 -> wgt16[o][k*C + c] f16
// -------------------------------------------------------------------
__global__ __launch_bounds__(256) void dcn_pack_w(
    const float* __restrict__ wsrc, _Float16* __restrict__ wdst)
{
    int tid = blockIdx.x * blockDim.x + threadIdx.x;
    if (tid >= OUTC * CK) return;
    int o  = tid / CK;
    int ck = tid % CK;          // dst index: k*C + c
    int k  = ck / CH;
    int c  = ck % CH;
    wdst[tid] = (_Float16)wsrc[((size_t)o * CH + c) * K9 + k];
}

// -------------------------------------------------------------------
// Fragment load: 16-bit operand, 16x32 layout (ISA 7.12.2):
//   lanes 0-15  hold K = {0..7, 16..23}
//   lanes 16-31 hold K = {8..15, 24..31}
// base points at rowStart; laneHi = 8*(lane>>4).
// -------------------------------------------------------------------
__device__ inline v16h load_frag16(const _Float16* __restrict__ base, int laneHi)
{
    v8h lo = *(const v8h*)(base + laneHi);        // K = laneHi .. +7
    v8h hi = *(const v8h*)(base + laneHi + 16);   // K = laneHi+16 .. +23
    return __builtin_shufflevector(lo, hi, 0,1,2,3,4,5,6,7,8,9,10,11,12,13,14,15);
}

// -------------------------------------------------------------------
// Kernel 3: GEMM  out[b][o][s] = sum_ck wgt16[o][ck]*colT[b][s][ck] + bias[o]
// block = 256 threads (8 waves). Wave tile = 32(M) x 64(N):
//   2 (M) x 4 (N) = 8 v_wmma per 32-wide K step, register double-buffered
//   so the next step's 12 b128 loads are in flight during the WMMAs.
// Block tile = 128(M) x 128(N): waves arranged 4 in M x 2 in N.
// grid = (O/128, S/128, B) = (2, 72, 2)
// -------------------------------------------------------------------
#define MT 2   // M tiles per wave
#define NT 4   // N tiles per wave

struct FragSet { v16h a[MT]; v16h b[NT]; };

__device__ inline void load_set(FragSet& f,
                                const _Float16* const* ap,
                                const _Float16* const* bp,
                                int kk, int laneHi)
{
    #pragma unroll
    for (int i = 0; i < MT; ++i) f.a[i] = load_frag16(ap[i] + kk, laneHi);
    #pragma unroll
    for (int t = 0; t < NT; ++t) f.b[t] = load_frag16(bp[t] + kk, laneHi);
}

__device__ inline void mma_set(const FragSet& f, v8f acc[MT][NT])
{
    #pragma unroll
    for (int i = 0; i < MT; ++i)
        #pragma unroll
        for (int t = 0; t < NT; ++t)
            acc[i][t] = __builtin_amdgcn_wmma_f32_16x16x32_f16(
                false, f.a[i], false, f.b[t], (short)0, acc[i][t], false, false);
}

__global__ __launch_bounds__(256) void dcn_gemm_wmma(
    const _Float16* __restrict__ wgt,   // [O][CK]
    const _Float16* __restrict__ colT,  // [B][S][CK]
    const float*    __restrict__ bias,  // [O]
    float*          __restrict__ out)   // [B][O][S]
{
    const int lane   = threadIdx.x & 31;
    const int wave   = threadIdx.x >> 5;
    const int waveM  = wave & 3;          // 4 waves in M
    const int waveN  = wave >> 2;         // 2 waves in N
    const int b      = blockIdx.z;
    const int m0     = blockIdx.x * 128 + waveM * (16 * MT);
    const int n0     = blockIdx.y * 128 + waveN * (16 * NT);
    const int laneHi = (lane >> 4) * 8;   // K-chunk select; also C-matrix M half offset
    const int lcol   = lane & 15;

    const _Float16* ap[MT];
    const _Float16* bp[NT];
    #pragma unroll
    for (int i = 0; i < MT; ++i)
        ap[i] = wgt + (size_t)(m0 + i * 16 + lcol) * CK;
    #pragma unroll
    for (int t = 0; t < NT; ++t)
        bp[t] = colT + ((size_t)b * SPIX + n0 + t * 16 + lcol) * CK;

    v8f acc[MT][NT];
    #pragma unroll
    for (int i = 0; i < MT; ++i)
        #pragma unroll
        for (int t = 0; t < NT; ++t) {
            v8f z = {0,0,0,0,0,0,0,0};
            acc[i][t] = z;
        }

    // software pipeline: 2 k-steps per iteration, ping-pong fragment sets
    FragSet f0, f1;
    load_set(f0, ap, bp, 0, laneHi);
    #pragma unroll 1
    for (int kk = 0; kk < CK - 64; kk += 64) {
        load_set(f1, ap, bp, kk + 32, laneHi);
        mma_set(f0, acc);
        load_set(f0, ap, bp, kk + 64, laneHi);
        mma_set(f1, acc);
    }
    load_set(f1, ap, bp, CK - 32, laneHi);
    mma_set(f0, acc);   // k-step CK-64
    mma_set(f1, acc);   // k-step CK-32

    // C/D f32 16x16 layout: VGPR r -> M = r + 8*(lane>=16), N = lane&15
    float* outp = out + (size_t)b * OUTC * SPIX;
    #pragma unroll
    for (int i = 0; i < MT; ++i) {
        #pragma unroll
        for (int t = 0; t < NT; ++t) {
            int n = n0 + t * 16 + lcol;
            #pragma unroll
            for (int r = 0; r < 8; ++r) {
                int row = m0 + i * 16 + r + laneHi;
                outp[(size_t)row * SPIX + n] = acc[i][t][r] + bias[row];
            }
        }
    }
}

// -------------------------------------------------------------------
extern "C" void kernel_launch(void* const* d_in, const int* in_sizes, int n_in,
                              void* d_out, int out_size, void* d_ws, size_t ws_size,
                              hipStream_t stream) {
    const float* input  = (const float*)d_in[0];
    const float* offset = (const float*)d_in[1];
    const float* mask   = (const float*)d_in[2];
    const float* weight = (const float*)d_in[3];
    const float* bias   = (const float*)d_in[4];
    float* out = (float*)d_out;

    // workspace partition (both 16B aligned)
    const size_t colBytes = (size_t)BATCH * SPIX * CK * sizeof(_Float16); // ~85 MB
    _Float16* colT  = (_Float16*)d_ws;
    _Float16* wgt16 = (_Float16*)((char*)d_ws + colBytes);                // ~1.2 MB

    {   // gather: one thread per (b,k,s)
        int total = BATCH * K9 * SPIX;           // 165888
        dcn_gather<<<(total + 255) / 256, 256, 0, stream>>>(input, offset, mask, colT);
    }
    {   // weight repack
        int total = OUTC * CK;                   // 589824
        dcn_pack_w<<<(total + 255) / 256, 256, 0, stream>>>(weight, wgt16);
    }
    {   // WMMA GEMM
        dim3 grid(OUTC / 128, SPIX / 128, BATCH); // (2,72,2)
        dcn_gemm_wmma<<<grid, 256, 0, stream>>>(wgt16, colT, bias, out);
    }
}
